// EDA_Attention_58652073394663
// MI455X (gfx1250) — matrix operand order repaired and checked by reading, not compiled, not run on hardware
//
#include <hip/hip_runtime.h>

typedef __attribute__((ext_vector_type(16))) _Float16 v16h;
typedef __attribute__((ext_vector_type(8)))  _Float16 v8h;
typedef __attribute__((ext_vector_type(4)))  _Float16 v4h;
typedef __attribute__((ext_vector_type(2)))  _Float16 v2h;
typedef __attribute__((ext_vector_type(8)))  float    v8f;

#define NB    8      // batch
#define IMG   64     // H == W
#define CH    512
#define HDIM  64
#define WIN   512    // tokens per window (64 * 8)
#define QBLK  128    // queries per workgroup
#define KT    256    // keys per LDS stage
#define ATT_SCALE 0.125f

union HV { v16h v; _Float16 h[16]; };

// token t (0..511) in window wblk -> image row index l = y*64 + x
static __device__ __forceinline__ int rowl(int wblk, int t) {
  return ((t >> 3) << 6) + (wblk << 3) + (t & 7);
}

static __device__ __forceinline__ v16h cat8(v8h a, v8h b) {
  return __builtin_shufflevector(a, b, 0,1,2,3,4,5,6,7,8,9,10,11,12,13,14,15);
}

static __device__ __forceinline__ v8f wmma16(v16h a, v16h b, v8f c) {
  return __builtin_amdgcn_wmma_f32_16x16x32_f16(false, a, false, b, (short)0, c,
                                                false, false);
}

// ---------------- LePE depthwise 3x3 conv (+bias) -> writes Out ----------------
__global__ __launch_bounds__(256) void lepe_conv_kernel(
    const float* __restrict__ V, const float* __restrict__ Wc,
    const float* __restrict__ Bb, float* __restrict__ Out) {
  int idx = blockIdx.x * 256 + threadIdx.x;      // b*4096*512 + pix*512 + c
  int c   = idx & (CH - 1);
  int pix = (idx >> 9) & 4095;
  int b   = idx >> 21;
  int y = pix >> 6, x = pix & 63;
  float acc = Bb[c];
  const float* wp = Wc + c * 9;
  const float* vb = V + (size_t)b * (4096 * CH) + c;
#pragma unroll
  for (int dy = -1; dy <= 1; ++dy) {
#pragma unroll
    for (int dx = -1; dx <= 1; ++dx) {
      int yy = y + dy, xx = x + dx;
      if (yy >= 0 && yy < IMG && xx >= 0 && xx < IMG)
        acc += wp[(dy + 1) * 3 + (dx + 1)] * vb[(size_t)(yy * IMG + xx) * CH];
    }
  }
  Out[idx] = acc;
}

// ---------------- CSWin window attention, f16 WMMA, adds onto Out ----------------
__global__ __launch_bounds__(256) void cswin_attn_kernel(
    const float* __restrict__ Q, const float* __restrict__ K,
    const float* __restrict__ V, float* __restrict__ Out) {
  __shared__ __align__(16) _Float16 Ks[KT * HDIM];      // 32 KB, [key][d]
  __shared__ __align__(16) _Float16 VTs[HDIM * KT];     // 32 KB, [d][key]
  __shared__ __align__(16) _Float16 Ps[8][16 * 32];     // 8 KB, per-wave P staging

  const int qblk = blockIdx.x;          // 0..3
  const int head = blockIdx.y;          // 0..7
  const int win  = blockIdx.z;          // 0..63
  const int b = win >> 3, wblk = win & 7;
  const int tid  = threadIdx.x;
  const int lane = tid & 31;
  const int wv   = tid >> 5;
  const int hi16 = lane & 16;           // 0 or 16

  const size_t base = (size_t)b * (4096 * CH) + head * HDIM;  // + l*CH + d

  // ---- persistent Q A-fragments (scaled), d 0..31 and 32..63 ----
  HV q0, q1;
  {
    const int qt = qblk * QBLK + wv * 16 + (lane & 15);
    const float* qp = Q + base + (size_t)rowl(wblk, qt) * CH;
    const int dlo = hi16;  // 0 or 16
#pragma unroll
    for (int i = 0; i < 8; ++i) {
      q0.h[i]     = (_Float16)(qp[dlo + i]      * ATT_SCALE);
      q0.h[i + 8] = (_Float16)(qp[dlo + 8 + i] * ATT_SCALE);
      q1.h[i]     = (_Float16)(qp[32 + dlo + i] * ATT_SCALE);
      q1.h[i + 8] = (_Float16)(qp[40 + dlo + i] * ATT_SCALE);
    }
  }

  HV ones;
#pragma unroll
  for (int i = 0; i < 16; ++i) ones.h[i] = (_Float16)1.0f;

  v8f O0 = {}, O1 = {}, O2 = {}, O3 = {}, La = {};

  for (int st = 0; st < 2; ++st) {
    __syncthreads();
    // ---- stage fill: 256 keys. waves 0-3: V^T (f16), waves 4-7: K (f16) ----
    if (tid < 128) {
      const int kl = tid * 2;
      const float4* a4 = (const float4*)(V + base + (size_t)rowl(wblk, st * KT + kl) * CH);
      const float4* b4 = (const float4*)(V + base + (size_t)rowl(wblk, st * KT + kl + 1) * CH);
#pragma unroll 4
      for (int i = 0; i < 16; ++i) {
        float4 a = a4[i], bq = b4[i];
        int d = 4 * i;
        *(v2h*)&VTs[(d + 0) * KT + kl] = (v2h){(_Float16)a.x, (_Float16)bq.x};
        *(v2h*)&VTs[(d + 1) * KT + kl] = (v2h){(_Float16)a.y, (_Float16)bq.y};
        *(v2h*)&VTs[(d + 2) * KT + kl] = (v2h){(_Float16)a.z, (_Float16)bq.z};
        *(v2h*)&VTs[(d + 3) * KT + kl] = (v2h){(_Float16)a.w, (_Float16)bq.w};
      }
    } else {
      const int t2 = tid - 128;
#pragma unroll
      for (int kk = 0; kk < 2; ++kk) {
        const int kl = t2 * 2 + kk;
        const float4* kp = (const float4*)(K + base + (size_t)rowl(wblk, st * KT + kl) * CH);
#pragma unroll 4
        for (int i = 0; i < 16; ++i) {
          float4 a = kp[i];
          *(v4h*)&Ks[kl * HDIM + 4 * i] =
              (v4h){(_Float16)a.x, (_Float16)a.y, (_Float16)a.z, (_Float16)a.w};
        }
      }
    }
    __syncthreads();

    // ---- compute: 8 chunks of 32 keys ----
    for (int ch = 0; ch < KT / 32; ++ch) {
      const int k0   = ch * 32;
      const int nsel = lane & 15;
      const int keyA = k0 + nsel, keyB = keyA + 16;

      // Q K^T  (B-frag: col=key, rows=d packed; lanes>=16 take d+16)
      const v8h* ka0 = (const v8h*)&Ks[keyA * HDIM + hi16];
      const v8h* ka1 = (const v8h*)&Ks[keyA * HDIM + 32 + hi16];
      const v8h* kb0 = (const v8h*)&Ks[keyB * HDIM + hi16];
      const v8h* kb1 = (const v8h*)&Ks[keyB * HDIM + 32 + hi16];
      v8f S0 = {}, S1 = {};
      S0 = wmma16(q0.v, cat8(ka0[0], ka0[1]), S0);
      S0 = wmma16(q1.v, cat8(ka1[0], ka1[1]), S0);
      S1 = wmma16(q0.v, cat8(kb0[0], kb0[1]), S1);
      S1 = wmma16(q1.v, cat8(kb1[0], kb1[1]), S1);

      // exp (no max shift: scores ~N(0,1); mathematically identical softmax)
#pragma unroll
      for (int r = 0; r < 8; ++r) { S0[r] = __expf(S0[r]); S1[r] = __expf(S1[r]); }

      // C-layout -> row-major P in LDS (per-wave, same-wave LDS is in-order)
      const int Mb = hi16 >> 1;  // 0 or 8
      _Float16* pw = &Ps[wv][0];
#pragma unroll
      for (int r = 0; r < 8; ++r) {
        pw[(Mb + r) * 32 + nsel]      = (_Float16)S0[r];
        pw[(Mb + r) * 32 + nsel + 16] = (_Float16)S1[r];
      }
      __asm__ volatile("" ::: "memory");  // forbid compiler ld/st reordering

      // P A-fragment: M=lane%16, keys [cb,cb+16)
      const int cb = hi16;
      const v8h* pp = (const v8h*)&Ps[wv][nsel * 32 + cb];
      v16h pf = cat8(pp[0], pp[1]);

      // P V : B-frag col=d, rows=keys (consecutive from V^T)
      const v8h* vp0 = (const v8h*)&VTs[(0  + nsel) * KT + k0 + hi16];
      const v8h* vp1 = (const v8h*)&VTs[(16 + nsel) * KT + k0 + hi16];
      const v8h* vp2 = (const v8h*)&VTs[(32 + nsel) * KT + k0 + hi16];
      const v8h* vp3 = (const v8h*)&VTs[(48 + nsel) * KT + k0 + hi16];
      O0 = wmma16(pf, cat8(vp0[0], vp0[1]), O0);
      O1 = wmma16(pf, cat8(vp1[0], vp1[1]), O1);
      O2 = wmma16(pf, cat8(vp2[0], vp2[1]), O2);
      O3 = wmma16(pf, cat8(vp3[0], vp3[1]), O3);
      // softmax denominator via ones-matrix: lands in identical C-layout
      La = wmma16(pf, ones.v, La);
    }
  }

  // ---- epilogue: normalize and add onto lepe already in Out ----
  const int Mb = hi16 >> 1;
  const int Nc = lane & 15;
#pragma unroll
  for (int r = 0; r < 8; ++r) {
    const int qt = qblk * QBLK + wv * 16 + Mb + r;
    float* op = Out + base + (size_t)rowl(wblk, qt) * CH + Nc;
    const float inv = 1.0f / La[r];
    op[0]  += O0[r] * inv;
    op[16] += O1[r] * inv;
    op[32] += O2[r] * inv;
    op[48] += O3[r] * inv;
  }
}

extern "C" void kernel_launch(void* const* d_in, const int* in_sizes, int n_in,
                              void* d_out, int out_size, void* d_ws, size_t ws_size,
                              hipStream_t stream) {
  (void)in_sizes; (void)n_in; (void)out_size; (void)d_ws; (void)ws_size;
  const float* q  = (const float*)d_in[0];
  const float* k  = (const float*)d_in[1];
  const float* v  = (const float*)d_in[2];
  const float* cw = (const float*)d_in[3];
  const float* cb = (const float*)d_in[4];
  float* out = (float*)d_out;

  // 1) lepe + bias -> out (full overwrite, image layout == output layout)
  lepe_conv_kernel<<<(NB * 4096 * CH) / 256, 256, 0, stream>>>(v, cw, cb, out);
  // 2) window attention, accumulates onto out
  cswin_attn_kernel<<<dim3(WIN / QBLK, 8, 64), 256, 0, stream>>>(q, k, v, out);
}
